// DynamicMoE_22239340659019
// MI455X (gfx1250) — matrix operand order repaired and checked by reading, not compiled, not run on hardware
//
#include <hip/hip_runtime.h>
#include <hip/hip_bf16.h>

// Problem constants (from reference): B=4, S=2048 -> T=8192; D=1024; H=4096; E=8
#define T_TOK 8192
#define DDIM  1024
#define HDIM  4096
#define NEXP  8

typedef __attribute__((ext_vector_type(16))) __bf16       v16bf;
typedef __attribute__((ext_vector_type(8)))  float        v8f;
typedef __attribute__((ext_vector_type(4)))  float        f32x4;
typedef __attribute__((ext_vector_type(4)))  unsigned int u32x4;
typedef __attribute__((ext_vector_type(4)))  __bf16       bf16x4;

union FragBF { u32x4 u[2]; v16bf v; };   // 32B: one 16x32-bf16 A/B fragment per lane

// ---------------------------------------------------------------- utilities
__global__ void k_cvt_bf16(const float* __restrict__ x, __bf16* __restrict__ y, int n4) {
    int i = blockIdx.x * blockDim.x + threadIdx.x;
    if (i < n4) {
        f32x4 f = ((const f32x4*)x)[i];
        ((bf16x4*)y)[i] = __builtin_convertvector(f, bf16x4);
    }
}

__global__ void k_zero_f32(float* __restrict__ out, int n4) {
    int i = blockIdx.x * blockDim.x + threadIdx.x;
    if (i < n4) ((f32x4*)out)[i] = (f32x4)0.0f;
}

// Tiled transpose + fp32->bf16 convert:  W[K][N] f32  ->  Wt[N][K] bf16
__global__ __launch_bounds__(256)
void k_transpose_cvt(const float* __restrict__ W, __bf16* __restrict__ Wt,
                     int K, int N) {
    __shared__ float tile[32][33];           // +1 pad: no bank conflicts
    const int kb = blockIdx.y * 32, nb = blockIdx.x * 32;
    const int tx = threadIdx.x & 31, ty = threadIdx.x >> 5;   // 32x8
#pragma unroll
    for (int i = ty; i < 32; i += 8)
        tile[i][tx] = W[(size_t)(kb + i) * N + nb + tx];
    __syncthreads();
#pragma unroll
    for (int i = ty; i < 32; i += 8)
        Wt[(size_t)(nb + i) * K + kb + tx] = (__bf16)tile[tx][i];
}

// gating: w[t,e] = p > 0.5 ? p : 0,  p = sigmoid(x[t]·Wg[:,e] + bg[e])
__global__ void k_gate(const float* __restrict__ x, const float* __restrict__ Wg,
                       const float* __restrict__ bg, float* __restrict__ w) {
    int gid = blockIdx.x * blockDim.x + threadIdx.x;
    if (gid >= T_TOK * NEXP) return;
    int t = gid / NEXP, e = gid % NEXP;
    const float* xr = x + (size_t)t * DDIM;
    float acc = bg[e];
    for (int d = 0; d < DDIM; ++d) acc = fmaf(xr[d], Wg[d * NEXP + e], acc);
    float p = 1.0f / (1.0f + __expf(-acc));
    w[gid] = (p > 0.5f) ? p : 0.0f;
}

// ---------------------------------------------------------------- core GEMM
// acc[4][4] (+=) A[128 rows at m0][K] x Bt[256 rows at n0][K]^T
// A row-major [M][K] bf16, Bt row-major [N][K] bf16 (B transposed).
// Block = 256 threads = 8 waves in a 2(M) x 4(N) grid; wave tile 64x64
// (4x4 WMMA fragments -> 1.0 ds_load_b128 per WMMA).
// Double-buffered LDS (2 x (8KB A + 16KB B)); one barrier per K-step.
// NOTE: all LDS accesses index the __shared__ arrays directly so the
// compiler keeps addrspace(3) ds_load/ds_store (no generic/flat pointers).
__device__ __forceinline__ void wmma_gemm_128x256(
    const __bf16* __restrict__ A, int lda,
    const __bf16* __restrict__ Bt, int ldb,
    int K, int m0, int n0, v8f (&acc)[4][4])
{
    __shared__ alignas(16) __bf16 sA[2][128 * 32];
    __shared__ alignas(16) __bf16 sB[2][256 * 32];

    const int tid  = threadIdx.x;
    const int lane = tid & 31, wave = tid >> 5;
    const int wm = (wave >> 2) * 64, wn = (wave & 3) * 64;
    const int lrow = lane & 15, lg = lane >> 4;

    // staging: each thread copies one 32B segment of A and two of Bt
    const int r = tid >> 1, s = (tid & 1) * 16;
    const int stA  = r * 32 + s;           // LDS element offsets (per buffer)
    const int stB0 = r * 32 + s;
    const int stB1 = (r + 128) * 32 + s;
    const __bf16* gA  = A  + (size_t)(m0 + r) * lda + s;
    const __bf16* gB0 = Bt + (size_t)(n0 + r) * ldb + s;
    const __bf16* gB1 = gB0 + (size_t)128 * ldb;

    {   // preload tile 0
        const u32x4* a  = (const u32x4*)gA;
        const u32x4* b0 = (const u32x4*)gB0;
        const u32x4* b1 = (const u32x4*)gB1;
        u32x4* pa  = (u32x4*)&sA[0][stA];
        u32x4* pb0 = (u32x4*)&sB[0][stB0];
        u32x4* pb1 = (u32x4*)&sB[0][stB1];
        pa[0]  = a[0];  pa[1]  = a[1];
        pb0[0] = b0[0]; pb0[1] = b0[1];
        pb1[0] = b1[0]; pb1[1] = b1[1];
    }
    __syncthreads();

    int cur = 0;
    for (int k0 = 0; k0 < K; k0 += 32, cur ^= 1) {
        const bool more = (k0 + 32) < K;
        u32x4 ra0, ra1, rb00, rb01, rb10, rb11;
        if (more) {   // issue next tile's global loads; latency hides under WMMA
            const u32x4* a  = (const u32x4*)(gA  + k0 + 32);
            const u32x4* b0 = (const u32x4*)(gB0 + k0 + 32);
            const u32x4* b1 = (const u32x4*)(gB1 + k0 + 32);
            ra0  = a[0];  ra1  = a[1];
            rb00 = b0[0]; rb01 = b0[1];
            rb10 = b1[0]; rb11 = b1[1];
            __builtin_prefetch(gA  + k0 + 64, 0, 3);   // global_prefetch_b8
            __builtin_prefetch(gB0 + k0 + 64, 0, 3);
            __builtin_prefetch(gB1 + k0 + 64, 0, 3);
        }

        // A fragments (four M sub-tiles), per documented 16-bit A layout:
        // lane half h<8 -> K = 8g+h ; h>=8 -> K = 16+8g+(h-8)
        FragBF af[4];
#pragma unroll
        for (int i = 0; i < 4; ++i) {
            const int ap = (wm + i * 16 + lrow) * 32;
            af[i].u[0] = *(const u32x4*)&sA[cur][ap + lg * 8];
            af[i].u[1] = *(const u32x4*)&sA[cur][ap + 16 + lg * 8];
        }

        // B fragments: column N = lane&15, K = 8g..8g+7 and 16+8g..16+8g+7 (same as A).
        // Software-pipelined: load b[j+1] before the WMMA quad on b[j].
        FragBF bcur, bnxt;
        {
            const int bp = (wn + lrow) * 32 + lg * 8;
            bcur.u[0] = *(const u32x4*)&sB[cur][bp];
            bcur.u[1] = *(const u32x4*)&sB[cur][bp + 16];
        }
#pragma unroll
        for (int j = 0; j < 4; ++j) {
            if (j < 3) {
                const int bp = (wn + (j + 1) * 16 + lrow) * 32 + lg * 8;
                bnxt.u[0] = *(const u32x4*)&sB[cur][bp];
                bnxt.u[1] = *(const u32x4*)&sB[cur][bp + 16];
            }
#pragma unroll
            for (int i = 0; i < 4; ++i)
                acc[i][j] = __builtin_amdgcn_wmma_f32_16x16x32_bf16(
                    false, af[i].v, false, bcur.v, (short)0, acc[i][j], false, false);
            bcur = bnxt;
        }

        if (more) {   // stage next tile into the other buffer (ds_store_b128)
            const int nxt = cur ^ 1;
            u32x4* pa  = (u32x4*)&sA[nxt][stA];
            u32x4* pb0 = (u32x4*)&sB[nxt][stB0];
            u32x4* pb1 = (u32x4*)&sB[nxt][stB1];
            pa[0]  = ra0;  pa[1]  = ra1;
            pb0[0] = rb00; pb0[1] = rb01;
            pb1[0] = rb10; pb1[1] = rb11;
        }
        __syncthreads();
    }
}

// ---------------------------------------------------------------- pass 1
// Hact[T,H] = relu(Xb[T,D] @ W1t[H,D]^T + b1e)   (bf16 out)
__global__ __launch_bounds__(256)
void k_ffn1(const __bf16* __restrict__ Xb, const __bf16* __restrict__ W1t,
            const float* __restrict__ b1e, __bf16* __restrict__ Hact) {
    const int m0 = blockIdx.y * 128, n0 = blockIdx.x * 256;
    v8f acc[4][4] = {};
    wmma_gemm_128x256(Xb, DDIM, W1t, DDIM, DDIM, m0, n0, acc);

    const int lane = threadIdx.x & 31, wave = threadIdx.x >> 5;
    const int wm = (wave >> 2) * 64, wn = (wave & 3) * 64;
    const int lrow = lane & 15, lg = lane >> 4;
#pragma unroll
    for (int i = 0; i < 4; ++i) {
        int Mbase = m0 + wm + i * 16 + lg * 8;
#pragma unroll
        for (int j = 0; j < 4; ++j) {
            int Ncol = n0 + wn + j * 16 + lrow;
            float bias = b1e[Ncol];
            __bf16* op = Hact + (size_t)Mbase * HDIM + Ncol;
#pragma unroll
            for (int v = 0; v < 8; ++v) {
                float val = acc[i][j][v] + bias;
                val = val > 0.0f ? val : 0.0f;
                op[(size_t)v * HDIM] = (__bf16)val;
            }
        }
    }
}

// ---------------------------------------------------------------- pass 2
// out[T,D] += w[:,e] * (Hact[T,H] @ W2t[D,H]^T + b2e)
__global__ __launch_bounds__(256)
void k_ffn2(const __bf16* __restrict__ Hact, const __bf16* __restrict__ W2t,
            const float* __restrict__ b2e, const float* __restrict__ wgate,
            float* __restrict__ out, int e) {
    const int m0 = blockIdx.y * 128, n0 = blockIdx.x * 256;
    v8f acc[4][4] = {};
    wmma_gemm_128x256(Hact, HDIM, W2t, HDIM, HDIM, m0, n0, acc);

    const int lane = threadIdx.x & 31, wave = threadIdx.x >> 5;
    const int wm = (wave >> 2) * 64, wn = (wave & 3) * 64;
    const int lrow = lane & 15, lg = lane >> 4;
#pragma unroll
    for (int i = 0; i < 4; ++i) {
        int Mbase = m0 + wm + i * 16 + lg * 8;
        float wv[8];
#pragma unroll
        for (int v = 0; v < 8; ++v) wv[v] = wgate[(size_t)(Mbase + v) * NEXP + e];
#pragma unroll
        for (int j = 0; j < 4; ++j) {
            int Ncol = n0 + wn + j * 16 + lrow;
            float bias = b2e[Ncol];
            float* op = out + (size_t)Mbase * DDIM + Ncol;
#pragma unroll
            for (int v = 0; v < 8; ++v) {
                if (wv[v] != 0.0f)
                    op[(size_t)v * DDIM] += wv[v] * (acc[i][j][v] + bias);
            }
        }
    }
}

// ---------------------------------------------------------------- launcher
extern "C" void kernel_launch(void* const* d_in, const int* in_sizes, int n_in,
                              void* d_out, int out_size, void* d_ws, size_t ws_size,
                              hipStream_t stream) {
    const float* x  = (const float*)d_in[0];
    const float* Wg = (const float*)d_in[1];
    const float* bg = (const float*)d_in[2];
    const float* W1 = (const float*)d_in[3];
    const float* b1 = (const float*)d_in[4];
    const float* W2 = (const float*)d_in[5];
    const float* b2 = (const float*)d_in[6];
    float* out = (float*)d_out;

    // workspace (~97 MB):
    //   [0,16MB)      xb    : x in bf16              [T, D]
    //   [16MB,+256KB) wgate : gate weights f32       [T, E]
    //   [17MB,81MB)   hact  : bf16 (current expert)  [T, H]
    //   [81MB,89MB)   w1t   : W1[e]^T bf16           [H, D]
    //   [89MB,97MB)   w2t   : W2[e]^T bf16           [D, H]
    char* ws = (char*)d_ws;
    __bf16* xb    = (__bf16*)ws;
    float*  wgate = (float*)(ws + (size_t)16 * 1024 * 1024);
    __bf16* hact  = (__bf16*)(ws + (size_t)17 * 1024 * 1024);
    __bf16* w1t   = (__bf16*)(ws + (size_t)81 * 1024 * 1024);
    __bf16* w2t   = (__bf16*)(ws + (size_t)89 * 1024 * 1024);

    const int n4 = T_TOK * DDIM / 4;
    k_cvt_bf16<<<(n4 + 255) / 256, 256, 0, stream>>>(x, xb, n4);
    k_zero_f32<<<(n4 + 255) / 256, 256, 0, stream>>>(out, n4);
    k_gate<<<(T_TOK * NEXP + 255) / 256, 256, 0, stream>>>(x, Wg, bg, wgate);

    dim3 gT1(HDIM / 32, DDIM / 32);     // transpose W1e: K=D, N=H
    dim3 gT2(DDIM / 32, HDIM / 32);     // transpose W2e: K=H, N=D
    dim3 g1(HDIM / 256, T_TOK / 128);   // 16 x 64 blocks
    dim3 g2(DDIM / 256, T_TOK / 128);   //  4 x 64 blocks
    for (int e = 0; e < NEXP; ++e) {
        k_transpose_cvt<<<gT1, 256, 0, stream>>>(W1 + (size_t)e * DDIM * HDIM, w1t, DDIM, HDIM);
        k_ffn1<<<g1, 256, 0, stream>>>(xb, w1t, b1 + (size_t)e * HDIM, hact);
        k_transpose_cvt<<<gT2, 256, 0, stream>>>(W2 + (size_t)e * HDIM * DDIM, w2t, HDIM, DDIM);
        k_ffn2<<<g2, 256, 0, stream>>>(hact, w2t, b2 + (size_t)e * DDIM, wgate, out, e);
    }
}
